// TriangleMultiplicativeUpdate_26783416058536
// MI455X (gfx1250) — compile-verified, matrix operand order checked by reading, not run on hardware
//
#include <hip/hip_runtime.h>
#include <hip/hip_bf16.h>

#define L 512
#define C 128
#define NPOS (L * L)  // 262144 positions

typedef __attribute__((ext_vector_type(16))) __bf16 v16bf;
typedef __attribute__((ext_vector_type(8)))  __bf16 v8bf;
typedef __attribute__((ext_vector_type(8)))  float  v8f;
typedef unsigned int u32x4 __attribute__((ext_vector_type(4)));
typedef int          i32x4 __attribute__((ext_vector_type(4)));
typedef int          i32x8 __attribute__((ext_vector_type(8)));

#if __has_builtin(__builtin_amdgcn_tensor_load_to_lds) && \
    __has_builtin(__builtin_amdgcn_s_wait_tensorcnt)
#define USE_TDM 1
#else
#define USE_TDM 0
#endif

// Build a 16-element bf16 A/B fragment for one lane.
// Per CDNA5 ISA (CDNA5 ISA), for 16-bit 16x32 A (and mirrored 32x16 B):
// lanes 0-15 hold K = {0..7, 16..23}, lanes 16-31 hold K = {8..15, 24..31}
// of each 32-wide K block. Caller folds the per-lane +8 offset (koff) into p.
static __device__ __forceinline__ v16bf frag_from(const __bf16* p) {
  v8bf lo = *(const v8bf*)(p);        // K rel: koff + 0..7
  v8bf hi = *(const v8bf*)(p + 16);   // K rel: koff + 16..23
  return __builtin_shufflevector(lo, hi, 0, 1, 2, 3, 4, 5, 6, 7,
                                          8, 9, 10, 11, 12, 13, 14, 15);
}

static __device__ __forceinline__ v8f wmma_bf16(v16bf a, v16bf b, v8f c) {
  return __builtin_amdgcn_wmma_f32_16x16x32_bf16(false, a, false, b,
                                                 (short)0, c, false, false);
}

static __device__ __forceinline__ float sigmoidf(float x) {
  return 1.0f / (1.0f + __expf(-x));
}

// ---------------------------------------------------------------------------
// Kernel 0: convert the 6 weight matrices (128x128 f32, row-major (out,in))
// to bf16. Row-major W is directly the WMMA B operand layout (lane n reads
// row W[n][k..] contiguously).  Order in Wb: pa, pb, ga, gb, out, g
// ---------------------------------------------------------------------------
__global__ void k_convert_weights(const float* __restrict__ wpa,
                                  const float* __restrict__ wpb,
                                  const float* __restrict__ wga,
                                  const float* __restrict__ wgb,
                                  const float* __restrict__ wout,
                                  const float* __restrict__ wg,
                                  __bf16* __restrict__ Wb) {
  int i = blockIdx.x * blockDim.x + threadIdx.x;  // 0 .. 6*16384-1
  int w = i >> 14;
  int off = i & 16383;
  const float* src;
  switch (w) {
    case 0: src = wpa; break;
    case 1: src = wpb; break;
    case 2: src = wga; break;
    case 3: src = wgb; break;
    case 4: src = wout; break;
    default: src = wg; break;
  }
  Wb[i] = (__bf16)src[off];
}

// ---------------------------------------------------------------------------
// Kernel 1: fused LN1 + 4 gated projections + output gate.
// One wave per 16-position tile; 4 waves / block. Writes aT/bT/gT in
// channel-major bf16 layout T[c][pos] so kernel 2/3 loads are contiguous.
// ---------------------------------------------------------------------------
__global__ void __launch_bounds__(128, 1)
k_ln_proj_gate(const float* __restrict__ z,
               const float* __restrict__ ln1_g, const float* __restrict__ ln1_b,
               const __bf16* __restrict__ Wb,
               const float* __restrict__ b_pa, const float* __restrict__ b_pb,
               const float* __restrict__ b_ga, const float* __restrict__ b_gb,
               const float* __restrict__ b_g,
               __bf16* __restrict__ aT, __bf16* __restrict__ bT,
               __bf16* __restrict__ gT) {
  const int lane = threadIdx.x & 31;
  const int wave = threadIdx.x >> 5;
  const int tileBase = (blockIdx.x * 4 + wave) * 16;  // 16 consecutive positions
  const int m = lane & 15;                            // row within tile
  const int koff = (lane & 16) ? 8 : 0;               // K-half per lane

  const float* zrow = z + (size_t)(tileBase + m) * C;

  // Load this lane's 64 of the row's 128 channels; accumulate stats.
  float zr[64];
  float sum = 0.f, ssq = 0.f;
#pragma unroll
  for (int kb = 0; kb < 4; ++kb) {
#pragma unroll
    for (int e = 0; e < 16; ++e) {
      int c = kb * 32 + ((e < 8) ? (koff + e) : (16 + koff + (e - 8)));
      float v = zrow[c];
      zr[kb * 16 + e] = v;
      sum += v;
      ssq += v * v;
    }
  }
  // Row halves live in lane l and lane l^16 (wave32): combine.
  sum += __shfl_xor(sum, 16, 32);
  ssq += __shfl_xor(ssq, 16, 32);
  float mean = sum * (1.0f / C);
  float var = ssq * (1.0f / C) - mean * mean;
  float rstd = rsqrtf(var + 1e-5f);

  // Normalize -> bf16 A fragments (16 pos x 128 K in 4 K-blocks).
  v16bf af[4];
#pragma unroll
  for (int kb = 0; kb < 4; ++kb) {
#pragma unroll
    for (int e = 0; e < 16; ++e) {
      int c = kb * 32 + ((e < 8) ? (koff + e) : (16 + koff + (e - 8)));
      float v = (zr[kb * 16 + e] - mean) * rstd * ln1_g[c] + ln1_b[c];
      af[kb][e] = (__bf16)v;
    }
  }

  const __bf16* Wpa = Wb;
  const __bf16* Wpb = Wb + 16384;
  const __bf16* Wga = Wb + 2 * 16384;
  const __bf16* Wgb = Wb + 3 * 16384;
  const __bf16* Wg = Wb + 5 * 16384;

  // a = sigmoid(z_norm@Wga^T)*(z_norm@Wpa^T);  b likewise with gb/pb.
#pragma unroll
  for (int which = 0; which < 2; ++which) {
    const __bf16* Wp = which ? Wpb : Wpa;
    const __bf16* Wgt = which ? Wgb : Wga;
    const float* bp = which ? b_pb : b_pa;
    const float* bg = which ? b_gb : b_ga;
    __bf16* outT = which ? bT : aT;
    for (int nt = 0; nt < 8; ++nt) {
      const int cn = nt * 16 + m;  // output channel for this lane's column
      v8f accp = {0.f, 0.f, 0.f, 0.f, 0.f, 0.f, 0.f, 0.f};
      v8f accg = {0.f, 0.f, 0.f, 0.f, 0.f, 0.f, 0.f, 0.f};
#pragma unroll
      for (int kb = 0; kb < 4; ++kb) {
        v16bf bpF = frag_from(Wp + cn * C + kb * 32 + koff);
        v16bf bgF = frag_from(Wgt + cn * C + kb * 32 + koff);
        accp = wmma_bf16(af[kb], bpF, accp);
        accg = wmma_bf16(af[kb], bgF, accg);
      }
      const float biasP = bp[cn], biasG = bg[cn];
      v8bf outv;
#pragma unroll
      for (int r = 0; r < 8; ++r)
        outv[r] = (__bf16)(sigmoidf(accg[r] + biasG) * (accp[r] + biasP));
      // Acc lane holds rows r + koff -> 8 consecutive positions: 16B store.
      *(v8bf*)(outT + (size_t)cn * NPOS + tileBase + koff) = outv;
    }
  }

  // Output gate g = sigmoid(z_norm @ Wg^T + b_g), stored channel-major bf16.
  for (int nt = 0; nt < 8; ++nt) {
    const int cn = nt * 16 + m;
    v8f acc = {0.f, 0.f, 0.f, 0.f, 0.f, 0.f, 0.f, 0.f};
#pragma unroll
    for (int kb = 0; kb < 4; ++kb)
      acc = wmma_bf16(af[kb], frag_from(Wg + cn * C + kb * 32 + koff), acc);
    const float bias = b_g[cn];
    v8bf outv;
#pragma unroll
    for (int r = 0; r < 8; ++r) outv[r] = (__bf16)sigmoidf(acc[r] + bias);
    *(v8bf*)(gT + (size_t)cn * NPOS + tileBase + koff) = outv;
  }
}

// ---------------------------------------------------------------------------
// Kernel 2: triangle einsum x[i,j,c] = sum_k a[i,k,c] * b[j,k,c].
// 128 independent 512x512x512 GEMMs (one per channel). Block = 256 threads
// (8 waves) computes one channel's 128x128 output block. K panels are moved
// by the Tensor Data Mover (tensor_load_to_lds, double-buffered, one issuing
// wave, s_wait_tensorcnt) with D#.pad inserting 16B per 64B row so the LDS
// image lands on a bank-conflict-free stride of KPAD=40 bf16.
// ---------------------------------------------------------------------------
#define KPAD 40  // 32 bf16 payload + 8 bf16 (16B) TDM pad per row

#if USE_TDM
// Issue one TDM 2D tile load: `rows` x 32 bf16 tile from a 512-wide row-major
// bf16 tensor (row stride 512 elements) at gaddr -> LDS ldsOff, padded.
// D# layout per CDNA5 ISA.
static __device__ __forceinline__ void tdm_load_2d(const __bf16* gaddr,
                                                   unsigned ldsOff, int rows) {
  unsigned long long ga = (unsigned long long)(size_t)(const void*)gaddr;
  u32x4 g0;
  g0.x = 1u;                                       // count=1 (valid user D#)
  g0.y = ldsOff;                                   // lds_addr (bytes)
  g0.z = (unsigned)(ga & 0xffffffffull);           // global_addr[31:0]
  g0.w = (unsigned)((ga >> 32) & 0x1ffffffull)     // global_addr[56:32]
         | (2u << 30);                             // type=2 (image)
  i32x8 g1;
  g1[0] = (1 << 16)      // data_size = 1 -> 2 bytes
          | (1 << 20)    // pad_enable
          | (3 << 22)    // pad_interval: every 64 bytes
          | (3 << 25);   // pad_amount: 4 DWORDs = 16 bytes
  g1[1] = (512 << 16);   // tensor_dim0 = 512 (lo16), atomic_barrier_addr = 0
  g1[2] = (512 << 16);   // tensor_dim0 hi = 0, tensor_dim1 = 512 (lo16)
  g1[3] = (32 << 16);    // tensor_dim1 hi = 0, tile_dim0 = 32
  g1[4] = rows;          // tile_dim1 = rows, tile_dim2 = 0
  g1[5] = 512;           // tensor_dim0_stride = 512 (lo32)
  g1[6] = 0;             // stride hi, tensor_dim1_stride lo
  g1[7] = 0;
  i32x4 z4 = {0, 0, 0, 0};
#if defined(__clang_major__) && __clang_major__ >= 23
  i32x8 z8 = {0, 0, 0, 0, 0, 0, 0, 0};
  __builtin_amdgcn_tensor_load_to_lds(g0, g1, z4, z4, z8, 0);
#else
  __builtin_amdgcn_tensor_load_to_lds(g0, g1, z4, z4, 0);
#endif
}
#endif  // USE_TDM

__global__ void __launch_bounds__(256, 1)
k_triangle(const __bf16* __restrict__ aT, const __bf16* __restrict__ bT,
           float* __restrict__ xT) {
  // [buffer][a=0/b=1][128 rows x KPAD]
  __shared__ __align__(16) __bf16 smem[2][2][128 * KPAD];

  const int jBase = blockIdx.x * 128;
  const int iBase = blockIdx.y * 128;
  const int ch = blockIdx.z;
  const __bf16* aC = aT + (size_t)ch * NPOS;
  const __bf16* bC = bT + (size_t)ch * NPOS;

  const int t = threadIdx.x;
  const int lane = t & 31;
  const int wave = t >> 5;          // 0..7 : 16-row strip of i
  const int m = lane & 15;
  const int koff = (lane & 16) ? 8 : 0;

  v8f acc[8];
#pragma unroll
  for (int tj = 0; tj < 8; ++tj)
    acc[tj] = (v8f){0.f, 0.f, 0.f, 0.f, 0.f, 0.f, 0.f, 0.f};

#if USE_TDM
  if (wave == 0) {  // one wave drives the Tensor Data Mover
    tdm_load_2d(aC + (size_t)iBase * L, (unsigned)(size_t)(void*)&smem[0][0][0],
                128);
    tdm_load_2d(bC + (size_t)jBase * L, (unsigned)(size_t)(void*)&smem[0][1][0],
                128);
  }
  for (int kc = 0; kc < 16; ++kc) {
    if (wave == 0) {
      if (kc < 15) {  // issue next chunk into the other buffer, then wait for
                      // the current chunk (<=2 outstanding = next chunk only)
        const int k0n = (kc + 1) * 32;
        tdm_load_2d(aC + (size_t)iBase * L + k0n,
                    (unsigned)(size_t)(void*)&smem[(kc + 1) & 1][0][0], 128);
        tdm_load_2d(bC + (size_t)jBase * L + k0n,
                    (unsigned)(size_t)(void*)&smem[(kc + 1) & 1][1][0], 128);
        __builtin_amdgcn_s_wait_tensorcnt(2);
      } else {
        __builtin_amdgcn_s_wait_tensorcnt(0);
      }
    }
    __syncthreads();  // current buffer now valid for all waves

    const __bf16* aBuf = &smem[kc & 1][0][0];
    const __bf16* bBuf = &smem[kc & 1][1][0];
    v16bf aF = frag_from(aBuf + (wave * 16 + m) * KPAD + koff);
#pragma unroll
    for (int tj = 0; tj < 8; ++tj) {
      v16bf bF = frag_from(bBuf + (tj * 16 + m) * KPAD + koff);
      acc[tj] = wmma_bf16(aF, bF, acc[tj]);
    }
  }
#else
  // Fallback: cooperative vector loads (no TDM builtin on this toolchain).
  const int ldRow = t >> 2;       // 0..63
  const int ldOff = (t & 3) * 8;  // 16B chunk
  for (int kc = 0; kc < 16; ++kc) {
    const int k0 = kc * 32;
    __syncthreads();
#pragma unroll
    for (int h = 0; h < 2; ++h) {
      int row = ldRow + h * 64;
      *(v8bf*)(&smem[0][0][0] + row * KPAD + ldOff) =
          *(const v8bf*)(aC + (size_t)(iBase + row) * L + k0 + ldOff);
      *(v8bf*)(&smem[0][1][0] + row * KPAD + ldOff) =
          *(const v8bf*)(bC + (size_t)(jBase + row) * L + k0 + ldOff);
    }
    __syncthreads();
    v16bf aF = frag_from(&smem[0][0][0] + (wave * 16 + m) * KPAD + koff);
#pragma unroll
    for (int tj = 0; tj < 8; ++tj) {
      v16bf bF = frag_from(&smem[0][1][0] + (tj * 16 + m) * KPAD + koff);
      acc[tj] = wmma_bf16(aF, bF, acc[tj]);
    }
  }
#endif

  // Store: xT[c][i*512+j], coalesced 64B across lanes 0-15 / 16-31.
  float* xC = xT + (size_t)ch * NPOS;
  const int i0 = iBase + wave * 16 + koff;  // acc rows r+koff
#pragma unroll
  for (int tj = 0; tj < 8; ++tj) {
#pragma unroll
    for (int r = 0; r < 8; ++r)
      xC[(size_t)(i0 + r) * L + jBase + tj * 16 + m] = acc[tj][r];
  }
}

// ---------------------------------------------------------------------------
// Kernel 3: LN2(x) + (x @ W_out^T + b_out) * gate + z  (fused epilogue).
// One wave per 16-position tile, like kernel 1; x read from channel-major xT.
// ---------------------------------------------------------------------------
__global__ void __launch_bounds__(128, 1)
k_ln2_out(const float* __restrict__ xT,
          const float* __restrict__ ln2_g, const float* __restrict__ ln2_b,
          const __bf16* __restrict__ Wb, const float* __restrict__ b_out,
          const __bf16* __restrict__ gT, const float* __restrict__ z,
          float* __restrict__ out) {
  const int lane = threadIdx.x & 31;
  const int wave = threadIdx.x >> 5;
  const int tileBase = (blockIdx.x * 4 + wave) * 16;
  const int m = lane & 15;
  const int koff = (lane & 16) ? 8 : 0;
  const int pos = tileBase + m;

  // Gather x for this row from channel-major layout (coalesced across lanes).
  float xr[64];
  float sum = 0.f, ssq = 0.f;
#pragma unroll
  for (int kb = 0; kb < 4; ++kb) {
#pragma unroll
    for (int e = 0; e < 16; ++e) {
      int c = kb * 32 + ((e < 8) ? (koff + e) : (16 + koff + (e - 8)));
      float v = xT[(size_t)c * NPOS + pos];
      xr[kb * 16 + e] = v;
      sum += v;
      ssq += v * v;
    }
  }
  sum += __shfl_xor(sum, 16, 32);
  ssq += __shfl_xor(ssq, 16, 32);
  float mean = sum * (1.0f / C);
  float var = ssq * (1.0f / C) - mean * mean;
  float rstd = rsqrtf(var + 1e-5f);

  v16bf xf[4];
#pragma unroll
  for (int kb = 0; kb < 4; ++kb) {
#pragma unroll
    for (int e = 0; e < 16; ++e) {
      int c = kb * 32 + ((e < 8) ? (koff + e) : (16 + koff + (e - 8)));
      float v = (xr[kb * 16 + e] - mean) * rstd * ln2_g[c] + ln2_b[c];
      xf[kb][e] = (__bf16)v;
    }
  }

  const __bf16* Wout = Wb + 4 * 16384;
  for (int nt = 0; nt < 8; ++nt) {
    const int cn = nt * 16 + m;
    v8f acc = {0.f, 0.f, 0.f, 0.f, 0.f, 0.f, 0.f, 0.f};
#pragma unroll
    for (int kb = 0; kb < 4; ++kb)
      acc = wmma_bf16(xf[kb], frag_from(Wout + cn * C + kb * 32 + koff), acc);
    const float bias = b_out[cn];
    // Gate stored channel-major: same (row,col) mapping as accumulator.
    v8bf gv = *(const v8bf*)(gT + (size_t)cn * NPOS + tileBase + koff);
#pragma unroll
    for (int r = 0; r < 8; ++r) {
      size_t idx = (size_t)(tileBase + koff + r) * C + cn;
      out[idx] = z[idx] + (float)gv[r] * (acc[r] + bias);
    }
  }
}

// ---------------------------------------------------------------------------
extern "C" void kernel_launch(void* const* d_in, const int* in_sizes, int n_in,
                              void* d_out, int out_size, void* d_ws,
                              size_t ws_size, hipStream_t stream) {
  (void)in_sizes; (void)n_in; (void)out_size; (void)ws_size;
  const float* z = (const float*)d_in[0];
  const float* ln1_g = (const float*)d_in[1];
  const float* ln1_b = (const float*)d_in[2];
  const float* W_pa = (const float*)d_in[3];
  const float* b_pa = (const float*)d_in[4];
  const float* W_pb = (const float*)d_in[5];
  const float* b_pb = (const float*)d_in[6];
  const float* W_ga = (const float*)d_in[7];
  const float* b_ga = (const float*)d_in[8];
  const float* W_gb = (const float*)d_in[9];
  const float* b_gb = (const float*)d_in[10];
  const float* W_out = (const float*)d_in[11];
  const float* b_out = (const float*)d_in[12];
  const float* W_g = (const float*)d_in[13];
  const float* b_g = (const float*)d_in[14];
  const float* ln2_g = (const float*)d_in[15];
  const float* ln2_b = (const float*)d_in[16];

  char* ws = (char*)d_ws;
  __bf16* aT = (__bf16*)ws;                              // 64 MB bf16
  __bf16* bT = (__bf16*)(ws + ((size_t)64 << 20));       // 64 MB bf16
  __bf16* gT = (__bf16*)(ws + ((size_t)128 << 20));      // 64 MB bf16
  float* xT = (float*)(ws + ((size_t)192 << 20));        // 128 MB f32
  __bf16* Wb = (__bf16*)(ws + ((size_t)320 << 20));      // 192 KB bf16 weights

  k_convert_weights<<<384, 256, 0, stream>>>(W_pa, W_pb, W_ga, W_gb, W_out, W_g,
                                             Wb);
  k_ln_proj_gate<<<4096, 128, 0, stream>>>(z, ln1_g, ln1_b, Wb, b_pa, b_pb,
                                           b_ga, b_gb, b_g, aT, bT, gT);
  k_triangle<<<dim3(4, 4, 128), 256, 0, stream>>>(aT, bT, xT);
  k_ln2_out<<<4096, 128, 0, stream>>>(xT, ln2_g, ln2_b, Wb, b_out, gT, z,
                                      (float*)d_out);
}
